// MoE_45011257262410
// MI455X (gfx1250) — matrix operand order repaired and checked by reading, not compiled, not run on hardware
//
#include <hip/hip_runtime.h>
#include <hip/hip_bf16.h>

// ---------------- problem constants ----------------
#define B_   4
#define N_   2048
#define D_   1024
#define E_   16
#define H_   4096
#define CAP  160              // max(min(N, int(N*1.25/E)), 4) = 160
#define T_   (B_*CAP)         // 640 rows per expert (=5*128, tiles exactly)

typedef __bf16 bf16;
typedef __attribute__((ext_vector_type(16))) __bf16 v16bf;
typedef __attribute__((ext_vector_type(8)))  __bf16 v8bf;
typedef __attribute__((ext_vector_type(8)))  float  v8f;

// =====================================================================
// 0) init: zero loss accumulators, slot table = -1 (ws is 0xAA-poisoned)
// =====================================================================
__global__ void moe_init_kernel(float* proxy, float* cnt1, int* slot) {
    int i = blockIdx.x * blockDim.x + threadIdx.x;
    if (i < B_ * E_) { proxy[i] = 0.0f; cnt1[i] = 0.0f; }
    for (int j = i; j < E_ * B_ * CAP; j += gridDim.x * blockDim.x) slot[j] = -1;
}

// =====================================================================
// 1) gating: one wave32 per token. logits -> softmax -> top2 -> gates
// =====================================================================
__global__ __launch_bounds__(256) void moe_gating_kernel(
    const float* __restrict__ x, const float* __restrict__ wg,
    const float* __restrict__ noise,
    int* __restrict__ idx1, int* __restrict__ idx2,
    float* __restrict__ g1o, float* __restrict__ g2o, int* __restrict__ k2o,
    float* __restrict__ proxy, float* __restrict__ cnt1)
{
    const int tok  = blockIdx.x * 8 + (threadIdx.x >> 5);   // B_*N_ waves total
    const int lane = threadIdx.x & 31;
    const float* xr = x + (size_t)tok * D_;

    float acc[E_] = {};
    for (int d = lane; d < D_; d += 32) {
        float xv = xr[d];
        const float4* wr = (const float4*)(wg + (size_t)d * E_);
        #pragma unroll
        for (int q = 0; q < 4; ++q) {
            float4 w4 = wr[q];
            acc[q*4+0] += xv * w4.x; acc[q*4+1] += xv * w4.y;
            acc[q*4+2] += xv * w4.z; acc[q*4+3] += xv * w4.w;
        }
    }
    #pragma unroll
    for (int e = 0; e < E_; ++e) {           // butterfly allreduce (wave32)
        float v = acc[e];
        #pragma unroll
        for (int off = 16; off; off >>= 1) v += __shfl_xor(v, off, 32);
        acc[e] = v;
    }
    // softmax (redundant on all lanes)
    float mx = acc[0];
    #pragma unroll
    for (int e = 1; e < E_; ++e) mx = fmaxf(mx, acc[e]);
    float s = 0.0f;
    #pragma unroll
    for (int e = 0; e < E_; ++e) { acc[e] = __expf(acc[e] - mx); s += acc[e]; }
    float inv = 1.0f / s;
    #pragma unroll
    for (int e = 0; e < E_; ++e) acc[e] *= inv;

    int i1 = 0; float b1 = acc[0];
    #pragma unroll
    for (int e = 1; e < E_; ++e) if (acc[e] > b1) { b1 = acc[e]; i1 = e; }
    int i2 = 0; float b2 = -1.0f;
    #pragma unroll
    for (int e = 0; e < E_; ++e) if (e != i1 && acc[e] > b2) { b2 = acc[e]; i2 = e; }

    float den = b1 + b2 + 1e-9f;
    float G1 = b1 / den, G2 = b2 / den;
    int kp2 = (noise[tok] < G2 / 0.2f) ? 1 : 0;

    if (lane == 0) {
        idx1[tok] = i1; idx2[tok] = i2;
        g1o[tok] = G1;  g2o[tok] = G2; k2o[tok] = kp2;
        int b = tok / N_;
        atomicAdd(&cnt1[b * E_ + i1], 1.0f);
        #pragma unroll
        for (int e = 0; e < E_; ++e) atomicAdd(&proxy[b * E_ + e], acc[e]);
    }
}

// =====================================================================
// 2) scan: sequential capacity cumsum per (b,e); builds slot->token table
// =====================================================================
__global__ void moe_scan_kernel(
    const int* __restrict__ idx1, const int* __restrict__ idx2,
    int* __restrict__ k2, int* __restrict__ keep1,
    int* __restrict__ pos1, int* __restrict__ pos2, int* __restrict__ slot)
{
    const int b = blockIdx.x;
    const int e = threadIdx.x;        // blockDim = 32, lanes 0..15 active
    __shared__ int cap[E_];
    if (e < E_) {
        int cnt = 0;
        for (int n = 0; n < N_; ++n) {
            int t = b * N_ + n;
            if (idx1[t] == e) {
                pos1[t] = cnt;
                int kp = (cnt < CAP);
                keep1[t] = kp;
                if (kp) slot[(e * B_ + b) * CAP + cnt] = n;
                ++cnt;
            }
        }
        cap[e] = (cnt < CAP) ? cnt : CAP;
    }
    __syncthreads();
    if (e < E_) {
        int c2 = 0, base = cap[e];
        for (int n = 0; n < N_; ++n) {
            int t = b * N_ + n;
            if (idx2[t] == e && k2[t]) {
                int p = base + c2;
                int kp = (p < CAP);
                k2[t] = kp; pos2[t] = p;
                if (kp) slot[(e * B_ + b) * CAP + p] = n;
                ++c2;
            }
        }
    }
}

// =====================================================================
// 3) gather: pack token rows into bf16 expert inputs [E][B*CAP][D]
// =====================================================================
__global__ __launch_bounds__(128) void moe_gather_kernel(
    const float* __restrict__ x, const int* __restrict__ slot, bf16* __restrict__ A)
{
    const int row = blockIdx.x;               // e*(B*CAP) + b*CAP + c
    const int tid = threadIdx.x;              // 128 threads, 8 elems each
    const int tok = slot[row];
    union { bf16 h[8]; uint4 u; } o;
    if (tok >= 0) {
        int b = (row / CAP) % B_;
        const float* s = x + ((size_t)b * N_ + tok) * D_ + tid * 8;
        float4 f0 = *(const float4*)s;
        float4 f1 = *(const float4*)(s + 4);
        o.h[0]=(bf16)f0.x; o.h[1]=(bf16)f0.y; o.h[2]=(bf16)f0.z; o.h[3]=(bf16)f0.w;
        o.h[4]=(bf16)f1.x; o.h[5]=(bf16)f1.y; o.h[6]=(bf16)f1.z; o.h[7]=(bf16)f1.w;
    } else {
        o.u = make_uint4(0u, 0u, 0u, 0u);
    }
    *(uint4*)(A + (size_t)row * D_ + tid * 8) = o.u;
}

// =====================================================================
// 4) batched expert GEMM: C[T x NTOT] = A_bf16[T x KTOT] * W_f32[KTOT x NTOT]
//    128x128 tile, double-buffered LDS, async A-tile (ASYNCcnt path),
//    pair-packed bf16 B staging, 8 waves * (2x4) v_wmma_f32_16x16x32_bf16.
// =====================================================================
template<int KTOT, int NTOT, bool RELU_BF16_OUT>
__global__ __launch_bounds__(256) void moe_gemm_kernel(
    const bf16* __restrict__ Aall, const float* __restrict__ Wall,
    bf16* __restrict__ Obf, float* __restrict__ Of)
{
    const int e  = blockIdx.z;
    const int m0 = blockIdx.y * 128;
    const int n0 = blockIdx.x * 128;
    const bf16*  A = Aall + (size_t)e * T_ * KTOT;
    const float* W = Wall + (size_t)e * KTOT * NTOT;

    __shared__ __align__(16) bf16 Al[2][128 * 32];   // [m][k]
    __shared__ __align__(32) bf16 Bl[2][128 * 32];   // [n][k] transposed

    const int tid  = threadIdx.x;
    const int lane = tid & 31;
    const int wid  = tid >> 5;
    const int wm   = (wid & 3) * 32;   // wave M offset (4 waves in M)
    const int wn   = (wid >> 2) * 64;  // wave N offset (2 waves in N)
    const int lm   = lane & 15;
    const int ka   = (lane >> 4) * 16; // A-frag k base (contiguous k, matches B)
    const int kb   = (lane >> 4) * 16; // B-frag k base (contiguous k, matches A)

    v8f acc[2][4] = {};
    float4 wreg[2][2];                 // staged W rows (pair of k rows)

    // --- async copy of A tile (bf16, pure byte move) into LDS ---
    auto issueA = [&](int k0, int buf) {
        #pragma unroll
        for (int j = 0; j < 2; ++j) {
            int seg = tid + j * 256;           // 512 x 16B segments
            int r = seg >> 2, c = (seg & 3) * 8;
            unsigned lds = (unsigned)(size_t)&Al[buf][r * 32 + c]; // addr[31:0]=LDS offset
            unsigned long long g =
                (unsigned long long)&A[(size_t)(m0 + r) * KTOT + k0 + c];
            asm volatile("global_load_async_to_lds_b128 %0, %1, off"
                         :: "v"(lds), "v"(g) : "memory");
        }
    };
    // --- W tile: load pair of k rows per thread into regs ---
    auto loadB = [&](int k0) {
        #pragma unroll
        for (int j = 0; j < 2; ++j) {
            int seg = tid + j * 256;           // 16 k-pairs x 32 col4 segments
            int kp = seg >> 5, c4 = seg & 31;
            const float* base = &W[(size_t)(k0 + kp * 2) * NTOT + n0 + c4 * 4];
            wreg[j][0] = *(const float4*)base;
            wreg[j][1] = *(const float4*)(base + NTOT);
        }
    };
    // --- convert+pack pairs, store as b32 into [n][k] LDS ---
    auto storeB = [&](int buf) {
        #pragma unroll
        for (int j = 0; j < 2; ++j) {
            int seg = tid + j * 256;
            int kp = seg >> 5, c4 = seg & 31;
            const float* lo = (const float*)&wreg[j][0];
            const float* hi = (const float*)&wreg[j][1];
            #pragma unroll
            for (int c = 0; c < 4; ++c) {
                union { bf16 h[2]; unsigned u; } pk;
                pk.h[0] = (bf16)lo[c];         // k = kp*2   (low 16)
                pk.h[1] = (bf16)hi[c];         // k = kp*2+1 (high 16)
                *(unsigned*)&Bl[buf][(c4 * 4 + c) * 32 + kp * 2] = pk.u;
            }
        }
    };

    // prologue: stage tile 0
    issueA(0, 0);
    loadB(0);
    storeB(0);
    asm volatile("s_wait_asynccnt 0" ::: "memory");
    __syncthreads();

    const int KITER = KTOT / 32;
    for (int it = 0; it < KITER; ++it) {
        const int cur = it & 1, nxt = cur ^ 1;
        const int k0 = it * 32;
        const bool more = (it + 1 < KITER);

        if (more) { issueA(k0 + 32, nxt); loadB(k0 + 32); }
        if (it + 2 < KITER)                // prefetch W two tiles ahead
            __builtin_prefetch(&W[(size_t)(k0 + 64 + (tid >> 3)) * NTOT + n0], 0, 0);

        v16bf bfrag[4];
        #pragma unroll
        for (int nj = 0; nj < 4; ++nj)
            bfrag[nj] = *(const v16bf*)&Bl[cur][(wn + nj * 16 + lm) * 32 + kb];

        #pragma unroll
        for (int mi = 0; mi < 2; ++mi) {
            int row = wm + mi * 16 + lm;
            union { v16bf v; v8bf h[2]; } af;
            af.h[0] = *(const v8bf*)&Al[cur][row * 32 + ka];
            af.h[1] = *(const v8bf*)&Al[cur][row * 32 + ka + 8];
            #pragma unroll
            for (int nj = 0; nj < 4; ++nj)
                acc[mi][nj] = __builtin_amdgcn_wmma_f32_16x16x32_bf16(
                    false, af.v, false, bfrag[nj], (short)0, acc[mi][nj],
                    false, false);
        }

        if (more) {
            storeB(nxt);                                   // other buffer: safe
            asm volatile("s_wait_asynccnt 0" ::: "memory");// A(nxt) landed
        }
        __syncthreads();
    }

    // ---- store: C/D layout  M = r + (lane>=16)*8, N = lane%16 ----
    const int mh = (lane >> 4) * 8;
    #pragma unroll
    for (int mi = 0; mi < 2; ++mi) {
        #pragma unroll
        for (int nj = 0; nj < 4; ++nj) {
            #pragma unroll
            for (int r = 0; r < 8; ++r) {
                int row = m0 + wm + mi * 16 + mh + r;
                int col = n0 + wn + nj * 16 + lm;
                float v = acc[mi][nj][r];
                if (RELU_BF16_OUT) {
                    Obf[((size_t)e * T_ + row) * NTOT + col] = (bf16)fmaxf(v, 0.0f);
                } else {
                    Of[((size_t)e * T_ + row) * NTOT + col] = v;
                }
            }
        }
    }
}

// =====================================================================
// 5) combine: out[b,n,:] = g1*Y[e1,b,c1,:] + g2*Y[e2,b,c2,:]
// =====================================================================
__global__ __launch_bounds__(128) void moe_combine_kernel(
    const float* __restrict__ Y,
    const int* __restrict__ idx1, const int* __restrict__ idx2,
    const float* __restrict__ g1, const float* __restrict__ g2,
    const int* __restrict__ keep1, const int* __restrict__ k2,
    const int* __restrict__ pos1, const int* __restrict__ pos2,
    float* __restrict__ out)
{
    const int t  = blockIdx.x;
    const int b  = t / N_;
    const int d0 = threadIdx.x * 8;
    float r[8] = {};
    if (keep1[t]) {
        float g = g1[t];
        const float* s = Y + ((size_t)idx1[t] * T_ + b * CAP + pos1[t]) * D_ + d0;
        #pragma unroll
        for (int i = 0; i < 8; ++i) r[i] += g * s[i];
    }
    if (k2[t]) {
        float g = g2[t];
        const float* s = Y + ((size_t)idx2[t] * T_ + b * CAP + pos2[t]) * D_ + d0;
        #pragma unroll
        for (int i = 0; i < 8; ++i) r[i] += g * s[i];
    }
    float* o = out + (size_t)t * D_ + d0;
    #pragma unroll
    for (int i = 0; i < 8; ++i) o[i] = r[i];
}

// =====================================================================
// 6) loss: LOSS_COEF * E^2 * mean_{b,e}( proxy/N * cnt1/N )
// =====================================================================
__global__ void moe_loss_kernel(const float* __restrict__ proxy,
                                const float* __restrict__ cnt1,
                                float* __restrict__ out_loss)
{
    int lane = threadIdx.x;   // 32
    float s = 0.0f;
    for (int i = lane; i < B_ * E_; i += 32)
        s += (proxy[i] * (1.0f / N_)) * (cnt1[i] * (1.0f / N_));
    #pragma unroll
    for (int off = 16; off; off >>= 1) s += __shfl_xor(s, off, 32);
    if (lane == 0)
        *out_loss = s * (1.0f / (B_ * E_)) * (float)(E_ * E_) * 0.01f;
}

// =====================================================================
extern "C" void kernel_launch(void* const* d_in, const int* in_sizes, int n_in,
                              void* d_out, int out_size, void* d_ws, size_t ws_size,
                              hipStream_t stream)
{
    const float* x     = (const float*)d_in[0];   // [B,N,D]
    const float* wg    = (const float*)d_in[1];   // [D,E]
    const float* w1    = (const float*)d_in[2];   // [E,D,H]
    const float* w2    = (const float*)d_in[3];   // [E,H,D]
    const float* noise = (const float*)d_in[4];   // [B,N]
    float* out = (float*)d_out;                   // [B,N,D] ++ [1]

    // ---- workspace carving ----
    char* p = (char*)d_ws;
    auto carve = [&](size_t bytes) -> void* {
        void* r = (void*)p;
        p += (bytes + 255) & ~(size_t)255;
        return r;
    };
    const int TOK = B_ * N_;
    int*   idx1  = (int*)  carve(TOK * 4);
    int*   idx2  = (int*)  carve(TOK * 4);
    float* g1    = (float*)carve(TOK * 4);
    float* g2    = (float*)carve(TOK * 4);
    int*   k2    = (int*)  carve(TOK * 4);
    int*   keep1 = (int*)  carve(TOK * 4);
    int*   pos1  = (int*)  carve(TOK * 4);
    int*   pos2  = (int*)  carve(TOK * 4);
    float* proxy = (float*)carve(B_ * E_ * 4);
    float* cnt1  = (float*)carve(B_ * E_ * 4);
    int*   slot  = (int*)  carve((size_t)E_ * B_ * CAP * 4);
    bf16*  Abf   = (bf16*) carve((size_t)E_ * T_ * D_ * 2);   // ~21 MB
    bf16*  H1    = (bf16*) carve((size_t)E_ * T_ * H_ * 2);   // ~84 MB
    float* Y     = (float*)carve((size_t)E_ * T_ * D_ * 4);   // ~42 MB

    // 0) init
    moe_init_kernel<<<40, 256, 0, stream>>>(proxy, cnt1, slot);
    // 1) gating: 8192 tokens, 8 waves/block
    moe_gating_kernel<<<TOK / 8, 256, 0, stream>>>(
        x, wg, noise, idx1, idx2, g1, g2, k2, proxy, cnt1);
    // 2) capacity scan: one block per batch
    moe_scan_kernel<<<B_, 32, 0, stream>>>(idx1, idx2, k2, keep1, pos1, pos2, slot);
    // 3) gather into bf16 expert inputs
    moe_gather_kernel<<<E_ * B_ * CAP, 128, 0, stream>>>(x, slot, Abf);
    // 4) GEMM1: H1 = relu(Abf @ w1)   [per expert: 640x1024 @ 1024x4096]
    moe_gemm_kernel<D_, H_, true><<<dim3(H_ / 128, T_ / 128, E_), 256, 0, stream>>>(
        Abf, w1, H1, nullptr);
    // 5) GEMM2: Y = H1 @ w2          [per expert: 640x4096 @ 4096x1024]
    moe_gemm_kernel<H_, D_, false><<<dim3(D_ / 128, T_ / 128, E_), 256, 0, stream>>>(
        H1, w2, nullptr, Y);
    // 6) combine back to tokens
    moe_combine_kernel<<<TOK, 128, 0, stream>>>(
        Y, idx1, idx2, g1, g2, keep1, k2, pos1, pos2, out);
    // 7) aux loss scalar
    moe_loss_kernel<<<1, 32, 0, stream>>>(proxy, cnt1, out + (size_t)TOK * D_);
}
